// MultiheadDotProductAttention_71227737637556
// MI455X (gfx1250) — matrix operand order repaired and checked by reading, not compiled, not run on hardware
//
#include <hip/hip_runtime.h>
#include <hip/hip_bf16.h>

typedef __bf16 bf16;
typedef bf16  v16bf __attribute__((ext_vector_type(16)));
typedef bf16  v8bf  __attribute__((ext_vector_type(8)));
typedef float v8f   __attribute__((ext_vector_type(8)));

#define B_  4
#define S_  2048
#define D_  1024
#define H_  16
#define HD_ 64
#define M_  (B_*S_)

static __device__ __forceinline__ v8f wmma_bf16(v16bf a, v16bf b, v8f c) {
  // D = A*B + C, 16x16x32 bf16 -> f32
  return __builtin_amdgcn_wmma_f32_16x16x32_bf16(false, a, false, b, (short)0, c, false, false);
}

// DPP16 lane permute within a 16-lane row (no LDS traffic, dual-issues with XDL).
template <int CTRL>
static __device__ __forceinline__ float dpp16(float v) {
  return __builtin_bit_cast(float,
      __builtin_amdgcn_update_dpp(0, __builtin_bit_cast(int, v),
                                  CTRL, 0xf, 0xf, true));
}
// Full reduction across a 16-lane row: quad xor1, quad xor2, half-mirror, mirror.
static __device__ __forceinline__ float rowmax16(float v) {
  v = fmaxf(v, dpp16<0xB1>(v));    // quad_perm [1,0,3,2]
  v = fmaxf(v, dpp16<0x4E>(v));    // quad_perm [2,3,0,1]
  v = fmaxf(v, dpp16<0x141>(v));   // row_half_mirror
  v = fmaxf(v, dpp16<0x140>(v));   // row_mirror
  return v;
}
static __device__ __forceinline__ float rowsum16(float v) {
  v += dpp16<0xB1>(v);
  v += dpp16<0x4E>(v);
  v += dpp16<0x141>(v);
  v += dpp16<0x140>(v);
  return v;
}

__global__ void cvt_f32_bf16(const float* __restrict__ src, bf16* __restrict__ dst, int n) {
  int i = blockIdx.x * blockDim.x + threadIdx.x;
  int stride = gridDim.x * blockDim.x;
  for (; i < n; i += stride) dst[i] = (bf16)src[i];
}

// C[m,n] = sum_k X[m,k] * W[n,k] + bias[n]   (torch Linear: x @ W.T + b)
// MODE 0: bf16 out, [B,H,S,HD]   (Q, K)
// MODE 1: bf16 out, [B,H,HD,S]   (V transposed, for contiguous P@V B-fragments)
// MODE 2: f32  out, row-major [M_, D_]  (final projection)
template <int MODE>
__global__ __launch_bounds__(128)
void gemm_bf16(const bf16* __restrict__ X, const bf16* __restrict__ W,
               const float* __restrict__ bias, bf16* __restrict__ dst_bf,
               float* __restrict__ dst_f) {
  const int lane = threadIdx.x & 31;
  const int wave = threadIdx.x >> 5;
  const int lh = lane >> 4;      // which 16-lane half
  const int lm = lane & 15;
  const int m0 = blockIdx.x * 256 + wave * 64;
  const int n0 = blockIdx.y * 64;

  v8f acc[4][4];
#pragma unroll
  for (int ni = 0; ni < 4; ++ni) {
    float bv = bias[n0 + ni * 16 + lm];
    v8f c;
#pragma unroll
    for (int r = 0; r < 8; ++r) c[r] = bv;
#pragma unroll
    for (int mi = 0; mi < 4; ++mi) acc[mi][ni] = c;
  }

  for (int k0 = 0; k0 < D_; k0 += 32) {
    v16bf af[4], bw[4];
    // A-fragment (16x32 bf16): lane holds K = lh*16..+15 of row lm
#pragma unroll
    for (int mi = 0; mi < 4; ++mi) {
      const bf16* ap = X + (size_t)(m0 + mi * 16 + lm) * D_ + k0 + lh * 16;
      v8bf a0 = *(const v8bf*)ap;
      v8bf a1 = *(const v8bf*)(ap + 8);
      v16bf a;
#pragma unroll
      for (int t = 0; t < 8; ++t) { a[t] = a0[t]; a[t + 8] = a1[t]; }
      af[mi] = a;
    }
    // B-fragment (32x16 bf16): lane holds 16 contiguous K from row n of W
#pragma unroll
    for (int ni = 0; ni < 4; ++ni) {
      const bf16* bp = W + (size_t)(n0 + ni * 16 + lm) * D_ + k0 + lh * 16;
      bw[ni] = *(const v16bf*)bp;
    }
#pragma unroll
    for (int mi = 0; mi < 4; ++mi)
#pragma unroll
      for (int ni = 0; ni < 4; ++ni)
        acc[mi][ni] = wmma_bf16(af[mi], bw[ni], acc[mi][ni]);
  }

  // Epilogue. All indexing that can be wave-uniform is hoisted to scalars:
  //  - a 64-wide n-tile lies inside exactly one head:  h  = n0>>6, hd = ni*16+lm
  //  - a 256-row m-block lies inside exactly one batch: b = m0>>11, s = (m0&2047)+...
  // C/D layout: VGPR r -> M = r (lanes 0-15) or r+8 (lanes 16-31); N = lane&15
  if (MODE == 2) {
    float* dp = dst_f + (size_t)m0 * D_ + n0;
#pragma unroll
    for (int mi = 0; mi < 4; ++mi)
#pragma unroll
      for (int r = 0; r < 8; ++r) {
        size_t rowoff = (size_t)(mi * 16 + r + 8 * lh) * D_;
#pragma unroll
        for (int ni = 0; ni < 4; ++ni)
          dp[rowoff + ni * 16 + lm] = acc[mi][ni][r];
      }
  } else {
    const int b = m0 >> 11;            // / S_
    const int h = n0 >> 6;             // / HD_
    const int sbase = m0 & (S_ - 1);
    if (MODE == 0) {
      bf16* dp = dst_bf + (((size_t)b * H_ + h) * S_ + sbase) * HD_;
#pragma unroll
      for (int mi = 0; mi < 4; ++mi)
#pragma unroll
        for (int r = 0; r < 8; ++r) {
          size_t rowoff = (size_t)(mi * 16 + r + 8 * lh) * HD_;
#pragma unroll
          for (int ni = 0; ni < 4; ++ni)
            dp[rowoff + ni * 16 + lm] = (bf16)acc[mi][ni][r];
        }
    } else {  // MODE 1: V transposed
      bf16* dp = dst_bf + (((size_t)b * H_ + h) * HD_) * S_ + sbase;
#pragma unroll
      for (int ni = 0; ni < 4; ++ni) {
        size_t coloff = (size_t)(ni * 16 + lm) * S_;
#pragma unroll
        for (int mi = 0; mi < 4; ++mi)
#pragma unroll
          for (int r = 0; r < 8; ++r)
            dp[coloff + mi * 16 + r + 8 * lh] = (bf16)acc[mi][ni][r];
      }
    }
  }
}

// Flash attention, one wave per 16-query tile, 32 keys per iteration.
// Q,K: [B,H,S,HD] bf16.  Vt: [B,H,HD,S] bf16.  Out: [B,S,D] bf16.
__global__ __launch_bounds__(256)
void attn_kernel(const bf16* __restrict__ Q, const bf16* __restrict__ K,
                 const bf16* __restrict__ Vt, bf16* __restrict__ Ob) {
  __shared__ bf16 sP[8][16][32];   // per-wave P tile, [query][key_local]
  const int lane = threadIdx.x & 31;
  const int wave = threadIdx.x >> 5;
  const int lh = lane >> 4, lm = lane & 15;
  const int bh = blockIdx.y;                 // b*H + h
  const int q0 = blockIdx.x * 128 + wave * 16;
  const float scale = 0.03125f;              // 1/sqrt(D_) = 1/32, exact

  const bf16* q_p = Q  + (size_t)bh * S_ * HD_;
  const bf16* k_p = K  + (size_t)bh * S_ * HD_;
  const bf16* v_p = Vt + (size_t)bh * S_ * HD_;

  // Q A-fragments covering hd 0..31 and 32..63
  v16bf qf[2];
#pragma unroll
  for (int c = 0; c < 2; ++c) {
    const bf16* ap = q_p + (size_t)(q0 + lm) * HD_ + c * 32 + lh * 16;
    v8bf a0 = *(const v8bf*)ap;
    v8bf a1 = *(const v8bf*)(ap + 8);
    v16bf a;
#pragma unroll
    for (int t = 0; t < 8; ++t) { a[t] = a0[t]; a[t + 8] = a1[t]; }
    qf[c] = a;
  }

  v8f o[4];
#pragma unroll
  for (int t = 0; t < 4; ++t)
#pragma unroll
    for (int r = 0; r < 8; ++r) o[t][r] = 0.0f;
  float mrow[8], lrow[8];
#pragma unroll
  for (int r = 0; r < 8; ++r) { mrow[r] = -3.0e38f; lrow[r] = 0.0f; }

  for (int kb = 0; kb < q0 + 16; kb += 32) {
    // ---- scores: two 16-key tiles, contracting over HD=64 (2 WMMAs each) ----
    v8f sc[2];
#pragma unroll
    for (int t = 0; t < 2; ++t) {
      v8f z;
#pragma unroll
      for (int r = 0; r < 8; ++r) z[r] = 0.0f;
      const bf16* bp = k_p + (size_t)(kb + t * 16 + lm) * HD_ + lh * 16;
      v16bf b0 = *(const v16bf*)bp;         // hd  0..31 slice
      v16bf b1 = *(const v16bf*)(bp + 32);  // hd 32..63 slice
      z = wmma_bf16(qf[0], b0, z);
      z = wmma_bf16(qf[1], b1, z);
      sc[t] = z;
    }
    // ---- causal mask (key > query -> -inf); wave-uniform: last chunk only ----
    if (kb + 31 > q0) {
#pragma unroll
      for (int t = 0; t < 2; ++t) {
        int key = kb + t * 16 + lm;
#pragma unroll
        for (int r = 0; r < 8; ++r) {
          int qq = q0 + r + 8 * lh;
          if (key > qq) sc[t][r] = -3.0e38f;
        }
      }
    }
    // ---- online softmax (per query row = VGPR r within a 16-lane half) ----
#pragma unroll
    for (int r = 0; r < 8; ++r) {
      float vmax = rowmax16(fmaxf(sc[0][r], sc[1][r]));
      float mnew = fmaxf(mrow[r], vmax);
      float alpha = __expf((mrow[r] - mnew) * scale);
      mrow[r] = mnew;
      lrow[r] *= alpha;
#pragma unroll
      for (int t = 0; t < 4; ++t) o[t][r] *= alpha;
      float p0 = __expf((sc[0][r] - mnew) * scale);
      float p1 = __expf((sc[1][r] - mnew) * scale);
      sc[0][r] = p0; sc[1][r] = p1;
      lrow[r] += rowsum16(p0 + p1);
    }
    // ---- P (C-layout) -> LDS row-major -> A-layout fragment ----
#pragma unroll
    for (int t = 0; t < 2; ++t)
#pragma unroll
      for (int r = 0; r < 8; ++r)
        sP[wave][r + 8 * lh][t * 16 + lm] = (bf16)sc[t][r];
    asm volatile("s_wait_dscnt 0" ::: "memory");
    v16bf pf;
    {
      const bf16* pp = &sP[wave][lm][lh * 16];
      v8bf a0 = *(const v8bf*)pp;
      v8bf a1 = *(const v8bf*)(pp + 8);
#pragma unroll
      for (int t = 0; t < 8; ++t) { pf[t] = a0[t]; pf[t + 8] = a1[t]; }
    }
    // ---- P @ V : 4 hd-tiles, K=32 keys contraction ----
#pragma unroll
    for (int t = 0; t < 4; ++t) {
      const bf16* vp = v_p + (size_t)(t * 16 + lm) * S_ + kb + lh * 16;
      v16bf vb = *(const v16bf*)vp;
      o[t] = wmma_bf16(pf, vb, o[t]);
    }
  }

  // ---- normalize, store to [B, S, D] bf16 ----
  int bidx = bh >> 4;           // / H_
  int h    = bh & (H_ - 1);
#pragma unroll
  for (int r = 0; r < 8; ++r) {
    float inv = 1.0f / lrow[r];
    int qq = q0 + r + 8 * lh;
#pragma unroll
    for (int t = 0; t < 4; ++t) {
      int hd = t * 16 + lm;
      Ob[((size_t)bidx * S_ + qq) * D_ + h * HD_ + hd] = (bf16)(o[t][r] * inv);
    }
  }
}

extern "C" void kernel_launch(void* const* d_in, const int* in_sizes, int n_in,
                              void* d_out, int out_size, void* d_ws, size_t ws_size,
                              hipStream_t stream) {
  (void)in_sizes; (void)n_in; (void)out_size; (void)ws_size;
  const float* x  = (const float*)d_in[0];
  const float* wq = (const float*)d_in[1];
  const float* bq = (const float*)d_in[2];
  const float* wk = (const float*)d_in[3];
  const float* bk = (const float*)d_in[4];
  const float* wv = (const float*)d_in[5];
  const float* bv = (const float*)d_in[6];
  const float* wo = (const float*)d_in[7];
  const float* bo = (const float*)d_in[8];
  float* out = (float*)d_out;

  const size_t nx = (size_t)M_ * D_;   // 8M elems
  const size_t nw = (size_t)D_ * D_;   // 1M elems
  char* p = (char*)d_ws;
  bf16* xb  = (bf16*)p; p += nx * 2;
  bf16* wqb = (bf16*)p; p += nw * 2;
  bf16* wkb = (bf16*)p; p += nw * 2;
  bf16* wvb = (bf16*)p; p += nw * 2;
  bf16* wob = (bf16*)p; p += nw * 2;
  bf16* Qb  = (bf16*)p; p += nx * 2;
  bf16* Kb  = (bf16*)p; p += nx * 2;
  bf16* Vtb = (bf16*)p; p += nx * 2;
  bf16* Ab  = (bf16*)p; p += nx * 2;   // attention output, [B,S,D] bf16

  cvt_f32_bf16<<<2048, 256, 0, stream>>>(x,  xb,  (int)nx);
  cvt_f32_bf16<<<512,  256, 0, stream>>>(wq, wqb, (int)nw);
  cvt_f32_bf16<<<512,  256, 0, stream>>>(wk, wkb, (int)nw);
  cvt_f32_bf16<<<512,  256, 0, stream>>>(wv, wvb, (int)nw);
  cvt_f32_bf16<<<512,  256, 0, stream>>>(wo, wob, (int)nw);

  dim3 gg(M_ / 256, D_ / 64);
  gemm_bf16<0><<<gg, 128, 0, stream>>>(xb, wqb, bq, Qb,  nullptr);
  gemm_bf16<0><<<gg, 128, 0, stream>>>(xb, wkb, bk, Kb,  nullptr);
  gemm_bf16<1><<<gg, 128, 0, stream>>>(xb, wvb, bv, Vtb, nullptr);

  dim3 ga(S_ / 128, B_ * H_);
  attn_kernel<<<ga, 256, 0, stream>>>(Qb, Kb, Vtb, Ab);

  gemm_bf16<2><<<gg, 128, 0, stream>>>(Ab, wob, bo, nullptr, out);
}
